// RotarySelfAttention_884763263728
// MI455X (gfx1250) — compile-verified, matrix operand order checked by reading, not run on hardware
//
#include <hip/hip_runtime.h>

// ---------------------------------------------------------------------------
// Types for CDNA5 WMMA (wave32, 16x16x32 bf16 -> f32)
// ---------------------------------------------------------------------------
typedef __attribute__((ext_vector_type(16))) __bf16 v16bf;
typedef __attribute__((ext_vector_type(8)))  float  v8f;
typedef __attribute__((ext_vector_type(4)))  unsigned int u32x4;
typedef __attribute__((ext_vector_type(4)))  int v4i;

union ABu  { u32x4 u[2]; v16bf v; };           // one WMMA A/B operand (16 bf16)
union U4H  { u32x4 q; unsigned short h[8]; };  // unpack 8 bf16 from a b128

static constexpr int cB  = 2;
static constexpr int cL  = 2048;
static constexpr int cD  = 2048;
static constexpr int cH  = 32;
static constexpr int cKV = 8;
static constexpr int cHD = 64;
static constexpr int cKVD = cKV * cHD;         // 512

__device__ __forceinline__ unsigned short f2bf(float f) {
    unsigned int u = __float_as_uint(f);
    u = (u + 0x7FFFu + ((u >> 16) & 1u)) >> 16;  // round-to-nearest-even
    return (unsigned short)u;
}

__device__ __forceinline__ v8f wmma_bf(ABu a, ABu b, v8f c) {
    return __builtin_amdgcn_wmma_f32_16x16x32_bf16(
        false, a.v, false, b.v, (short)0, c, false, false);
}

// ---------------------------------------------------------------------------
// Async global->LDS copy (CDNA5 ASYNCcnt path) with safe fallbacks
// ---------------------------------------------------------------------------
__device__ __forceinline__ void async_cp16(void* ldst, const void* gsrc) {
#if __has_builtin(__builtin_amdgcn_global_load_async_to_lds_b128)
    void* g = (void*)gsrc;  // drop const before addrspace cast
    __builtin_amdgcn_global_load_async_to_lds_b128(
        (__attribute__((address_space(1))) v4i*)g,
        (__attribute__((address_space(3))) v4i*)ldst, 0, 0);
#else
    *(u32x4*)ldst = *(const u32x4*)gsrc;
#endif
}

__device__ __forceinline__ void async_wait0() {
#if __has_builtin(__builtin_amdgcn_global_load_async_to_lds_b128)
#if __has_builtin(__builtin_amdgcn_s_wait_asynccnt)
    __builtin_amdgcn_s_wait_asynccnt(0);
#else
    asm volatile("s_wait_asynccnt 0x0" ::: "memory");
#endif
#endif
}

// ---------------------------------------------------------------------------
// 1) elementwise f32 -> bf16 conversion
// ---------------------------------------------------------------------------
__global__ void k_f32_to_bf16(const float* __restrict__ s,
                              unsigned short* __restrict__ d, int n) {
    int i = blockIdx.x * blockDim.x + threadIdx.x;
    if (i < n) d[i] = f2bf(s[i]);
}

// ---------------------------------------------------------------------------
// 2) bf16 GEMM: C(f32)[M,N] = A(bf16)[M,K] @ B(bf16)[K,N]
//    128x128 block tile, 256 threads (8 waves), each wave 32x64 (2x4 WMMA).
// ---------------------------------------------------------------------------
__global__ void k_gemm_bf16(const unsigned short* __restrict__ A,
                            const unsigned short* __restrict__ Bm,
                            float* __restrict__ C, int M, int N, int K) {
    __shared__ __align__(16) unsigned short ldsB[128][32];  // [n][k] transposed, 8KB

    const int tid  = threadIdx.x;
    const int lane = tid & 31;
    const int w    = tid >> 5;
    const int wm   = w >> 1;       // 0..3
    const int wn   = w & 1;        // 0..1
    const int lr   = lane & 15;    // row/col within 16
    const int hs   = lane >> 4;    // half-wave select
    const int kb   = hs * 8;       // k-chunk base inside operand
    const int blockN = blockIdx.x * 128;
    const int blockM = blockIdx.y * 128;

    v8f acc[2][4];
#pragma unroll
    for (int s = 0; s < 2; ++s)
#pragma unroll
        for (int t = 0; t < 4; ++t) acc[s][t] = (v8f){0,0,0,0,0,0,0,0};

    const int krow  = tid >> 3;          // 0..31
    const int nbase = (tid & 7) * 16;    // 0..112

    for (int k0 = 0; k0 < K; k0 += 32) {
        // cooperative load of B tile (32 x 128), stored transposed into LDS
        {
            const unsigned short* gp =
                Bm + (size_t)(k0 + krow) * N + blockN + nbase;
            U4H u0, u1;
            u0.q = *(const u32x4*)(gp);
            u1.q = *(const u32x4*)(gp + 8);
            if (k0 + 32 < K)
                __builtin_prefetch(Bm + (size_t)(k0 + 32 + krow) * N + blockN + nbase, 0, 0);
#pragma unroll
            for (int e = 0; e < 8; ++e) {
                ldsB[nbase + e][krow]     = u0.h[e];
                ldsB[nbase + 8 + e][krow] = u1.h[e];
            }
        }
        __syncthreads();

        // A operands straight from global (row-major matches A lane layout)
        ABu a[2];
#pragma unroll
        for (int s = 0; s < 2; ++s) {
            int m = blockM + wm * 32 + s * 16 + lr;
            const unsigned short* ap = A + (size_t)m * K + k0 + kb;
            a[s].u[0] = *(const u32x4*)(ap);
            a[s].u[1] = *(const u32x4*)(ap + 16);
        }
        // B operands from transposed LDS, 2x4 WMMA
#pragma unroll
        for (int t = 0; t < 4; ++t) {
            int n = wn * 64 + t * 16 + lr;
            ABu bb;
            bb.u[0] = *(const u32x4*)(&ldsB[n][kb]);
            bb.u[1] = *(const u32x4*)(&ldsB[n][kb + 16]);
#pragma unroll
            for (int s = 0; s < 2; ++s)
                acc[s][t] = wmma_bf(a[s], bb, acc[s][t]);
        }
        __syncthreads();
    }

    // epilogue: C layout — lane holds column lr, rows hs*8 + r
#pragma unroll
    for (int s = 0; s < 2; ++s)
#pragma unroll
        for (int t = 0; t < 4; ++t) {
            int n = blockN + wn * 64 + t * 16 + lr;
#pragma unroll
            for (int r = 0; r < 8; ++r) {
                int m = blockM + wm * 32 + s * 16 + hs * 8 + r;
                C[(size_t)m * N + n] = acc[s][t][r];
            }
        }
}

// ---------------------------------------------------------------------------
// 3) RoPE + RMSNorm per head row: one wave per (token, head)
// ---------------------------------------------------------------------------
__global__ void k_rope_rms(const float* __restrict__ src,
                           const float* __restrict__ cosb,
                           const float* __restrict__ sinb,
                           unsigned short* __restrict__ dst, int nh) {
    int idx  = blockIdx.x * 4 + (threadIdx.x >> 5);
    int lane = threadIdx.x & 31;
    int bl = idx / nh;
    int h  = idx % nh;
    int l  = bl % cL;

    const float* p = src + (size_t)bl * nh * cHD + h * cHD;
    float x1 = p[lane], x2 = p[lane + 32];
    float c = cosb[l * 32 + lane];
    float s = sinb[l * 32 + lane];
    float o1 = x1 * c - x2 * s;
    float o2 = x2 * c + x1 * s;

    float ss = o1 * o1 + o2 * o2;
#pragma unroll
    for (int m = 1; m <= 16; m <<= 1) ss += __shfl_xor(ss, m, 32);
    float r = rsqrtf(ss * (1.0f / 64.0f) + 1e-6f);

    unsigned short* q = dst + (size_t)bl * nh * cHD + h * cHD;
    q[lane]      = f2bf(o1 * r);
    q[lane + 32] = f2bf(o2 * r);
}

// ---------------------------------------------------------------------------
// 4) Flash attention: block = (b, h, 64-row Q tile), 4 waves x 16 Q rows.
//    K tile staged in LDS via async global->LDS (ASYNCcnt); V staged
//    transposed; QK^T + PV via v_wmma_f32_16x16x32_bf16; softmax in f32.
// ---------------------------------------------------------------------------
__global__ void k_attn(const unsigned short* __restrict__ qb,
                       const unsigned short* __restrict__ kbm,
                       const unsigned short* __restrict__ vbm,
                       unsigned short* __restrict__ aob) {
    __shared__ __align__(16) unsigned short ldsK[32][64];      // K tile (key x hd)
    __shared__ __align__(16) unsigned short ldsV[64][32];      // V^T tile (hd x key)
    __shared__ __align__(16) unsigned short ldsP[4][16][32];   // per-wave P tile

    const int bh  = blockIdx.x;
    const int b   = bh / cH;
    const int h   = bh % cH;
    const int kvh = h >> 2;                 // GQA group of 4
    const int qtile = blockIdx.y * 64;

    const int tid  = threadIdx.x;
    const int lane = tid & 31;
    const int w    = tid >> 5;
    const int lr   = lane & 15;
    const int hs   = lane >> 4;
    const int kb   = hs * 8;
    const int qrowbase = qtile + w * 16;

    // Q operands: row = lane&15, contiguous K chunks (A layout)
    ABu a0, a1;
    {
        int q = qrowbase + lr;
        const unsigned short* qp = qb + (size_t)(b * cL + q) * cD + h * cHD;
        a0.u[0] = *(const u32x4*)(qp + kb);
        a0.u[1] = *(const u32x4*)(qp + kb + 16);
        a1.u[0] = *(const u32x4*)(qp + kb + 32);
        a1.u[1] = *(const u32x4*)(qp + kb + 48);
    }

    float mrow[8], lrow[8];
    v8f acc[4];
#pragma unroll
    for (int r = 0; r < 8; ++r) { mrow[r] = -1e30f; lrow[r] = 0.0f; }
#pragma unroll
    for (int t = 0; t < 4; ++t) acc[t] = (v8f){0,0,0,0,0,0,0,0};

    const int vkey = tid >> 2;           // 0..31
    const int vhdb = (tid & 3) * 16;     // 0,16,32,48

    for (int j0 = 0; j0 < qtile + 64; j0 += 32) {
        // async copy of K tile (32 keys x 64 hd, row-major, no transpose needed)
        {
            int c0 = tid;            // chunk ids: 256 chunks of 8 bf16
            int c1 = tid + 128;
            int key0 = c0 >> 3, off0 = (c0 & 7) * 8;
            int key1 = c1 >> 3, off1 = (c1 & 7) * 8;
            const unsigned short* g0 =
                kbm + (size_t)(b * cL + j0 + key0) * cKVD + kvh * cHD + off0;
            const unsigned short* g1 =
                kbm + (size_t)(b * cL + j0 + key1) * cKVD + kvh * cHD + off1;
            async_cp16(&ldsK[key0][off0], g0);
            async_cp16(&ldsK[key1][off1], g1);
        }
        // cooperative transposed V tile load: ldsV[hd][key]
        {
            const unsigned short* vp =
                vbm + (size_t)(b * cL + j0 + vkey) * cKVD + kvh * cHD + vhdb;
            U4H v0, v1;
            v0.q = *(const u32x4*)(vp);
            v1.q = *(const u32x4*)(vp + 8);
#pragma unroll
            for (int e = 0; e < 8; ++e) {
                ldsV[vhdb + e][vkey]     = v0.h[e];
                ldsV[vhdb + 8 + e][vkey] = v1.h[e];
            }
        }
        async_wait0();
        __syncthreads();

        // scores for two 16-key subtiles (K operands from LDS)
        float s0[8], s1[8];
#pragma unroll
        for (int jj = 0; jj < 2; ++jj) {
            int klocal = jj * 16 + lr;
            int key = j0 + klocal;
            ABu b0, b1;
            b0.u[0] = *(const u32x4*)(&ldsK[klocal][kb]);
            b0.u[1] = *(const u32x4*)(&ldsK[klocal][kb + 16]);
            b1.u[0] = *(const u32x4*)(&ldsK[klocal][kb + 32]);
            b1.u[1] = *(const u32x4*)(&ldsK[klocal][kb + 48]);
            v8f sv = (v8f){0,0,0,0,0,0,0,0};
            sv = wmma_bf(a0, b0, sv);
            sv = wmma_bf(a1, b1, sv);
            float* dst = jj ? s1 : s0;
#pragma unroll
            for (int r = 0; r < 8; ++r) {
                int qr = qrowbase + hs * 8 + r;
                float v = sv[r] * 0.125f;               // 1/sqrt(64)
                dst[r] = (key <= qr) ? v : -10000.0f;   // causal mask
            }
        }

        // online softmax: row lives in one half-wave (16 lanes) at fixed r
#pragma unroll
        for (int r = 0; r < 8; ++r) {
            float t = fmaxf(s0[r], s1[r]);
#pragma unroll
            for (int m = 1; m <= 8; m <<= 1) t = fmaxf(t, __shfl_xor(t, m, 32));
            float mnew = fmaxf(mrow[r], t);
            float al = __expf(mrow[r] - mnew);
            mrow[r] = mnew;
            float p0 = __expf(s0[r] - mnew);
            float p1 = __expf(s1[r] - mnew);
            s0[r] = p0; s1[r] = p1;
            float rs = p0 + p1;
#pragma unroll
            for (int m = 1; m <= 8; m <<= 1) rs += __shfl_xor(rs, m, 32);
            lrow[r] = lrow[r] * al + rs;
#pragma unroll
            for (int t4 = 0; t4 < 4; ++t4) acc[t4][r] *= al;
        }

        // re-layout P (C layout -> A layout) through LDS
#pragma unroll
        for (int r = 0; r < 8; ++r) {
            int row = hs * 8 + r;
            ldsP[w][row][lr]      = f2bf(s0[r]);
            ldsP[w][row][16 + lr] = f2bf(s1[r]);
        }
        __syncthreads();

        ABu pa;
        pa.u[0] = *(const u32x4*)(&ldsP[w][lr][kb]);
        pa.u[1] = *(const u32x4*)(&ldsP[w][lr][kb + 16]);
#pragma unroll
        for (int t4 = 0; t4 < 4; ++t4) {
            int n = t4 * 16 + lr;
            ABu vb;
            vb.u[0] = *(const u32x4*)(&ldsV[n][kb]);
            vb.u[1] = *(const u32x4*)(&ldsV[n][kb + 16]);
            acc[t4] = wmma_bf(pa, vb, acc[t4]);
        }
        __syncthreads();
    }

    // normalize and store attention output (bf16, [b, q, h*64+hd])
#pragma unroll
    for (int t4 = 0; t4 < 4; ++t4)
#pragma unroll
        for (int r = 0; r < 8; ++r) {
            int qr = qrowbase + hs * 8 + r;
            float o = acc[t4][r] / lrow[r];
            aob[(size_t)(b * cL + qr) * cD + h * cHD + t4 * 16 + lr] = f2bf(o);
        }
}

// ---------------------------------------------------------------------------
// launch
// ---------------------------------------------------------------------------
extern "C" void kernel_launch(void* const* d_in, const int* in_sizes, int n_in,
                              void* d_out, int out_size, void* d_ws, size_t ws_size,
                              hipStream_t stream) {
    const float* x    = (const float*)d_in[0];
    const float* cosb = (const float*)d_in[1];
    const float* sinb = (const float*)d_in[2];
    const float* Wq   = (const float*)d_in[3];
    const float* Wk   = (const float*)d_in[4];
    const float* Wv   = (const float*)d_in[5];
    const float* Wo   = (const float*)d_in[6];
    float* out = (float*)d_out;

    const int M = cB * cL;   // 4096

    char* w = (char*)d_ws;
    size_t o = 0;
    unsigned short* xb  = (unsigned short*)(w + o); o += (size_t)M * cD * 2;
    unsigned short* Wqb = (unsigned short*)(w + o); o += (size_t)cD * cD * 2;
    unsigned short* Wkb = (unsigned short*)(w + o); o += (size_t)cD * cKVD * 2;
    unsigned short* Wvb = (unsigned short*)(w + o); o += (size_t)cD * cKVD * 2;
    unsigned short* Wob = (unsigned short*)(w + o); o += (size_t)cD * cD * 2;
    float* qf = (float*)(w + o); o += (size_t)M * cD * 4;
    float* kf = (float*)(w + o); o += (size_t)M * cKVD * 4;
    float* vf = (float*)(w + o); o += (size_t)M * cKVD * 4;
    unsigned short* qb2 = (unsigned short*)(w + o); o += (size_t)M * cD * 2;
    unsigned short* kb2 = (unsigned short*)(w + o); o += (size_t)M * cKVD * 2;
    unsigned short* vb2 = (unsigned short*)(w + o); o += (size_t)M * cKVD * 2;
    unsigned short* aob = (unsigned short*)(w + o); o += (size_t)M * cD * 2;

    // 1) bf16 conversions
    {
        int n;
        n = M * cD;       k_f32_to_bf16<<<(n + 255) / 256, 256, 0, stream>>>(x,  xb,  n);
        n = cD * cD;      k_f32_to_bf16<<<(n + 255) / 256, 256, 0, stream>>>(Wq, Wqb, n);
        n = cD * cKVD;    k_f32_to_bf16<<<(n + 255) / 256, 256, 0, stream>>>(Wk, Wkb, n);
        n = cD * cKVD;    k_f32_to_bf16<<<(n + 255) / 256, 256, 0, stream>>>(Wv, Wvb, n);
        n = cD * cD;      k_f32_to_bf16<<<(n + 255) / 256, 256, 0, stream>>>(Wo, Wob, n);
    }

    // 2) QKV projections (WMMA GEMM)
    k_gemm_bf16<<<dim3(cD   / 128, M / 128), 256, 0, stream>>>(xb, Wqb, qf, M, cD,   cD);
    k_gemm_bf16<<<dim3(cKVD / 128, M / 128), 256, 0, stream>>>(xb, Wkb, kf, M, cKVD, cD);
    k_gemm_bf16<<<dim3(cKVD / 128, M / 128), 256, 0, stream>>>(xb, Wvb, vf, M, cKVD, cD);

    // 3) RoPE + RMSNorm on Q/K; V straight to bf16
    k_rope_rms<<<(M * cH)  / 4, 128, 0, stream>>>(qf, cosb, sinb, qb2, cH);
    k_rope_rms<<<(M * cKV) / 4, 128, 0, stream>>>(kf, cosb, sinb, kb2, cKV);
    { int n = M * cKVD; k_f32_to_bf16<<<(n + 255) / 256, 256, 0, stream>>>(vf, vb2, n); }

    // 4) causal flash attention
    k_attn<<<dim3(cB * cH, cL / 64), 128, 0, stream>>>(qb2, kb2, vb2, aob);

    // 5) output projection -> f32 d_out
    k_gemm_bf16<<<dim3(cD / 128, M / 128), 256, 0, stream>>>(aob, Wob, out, M, cD, cD);
}
